// GraphFADE_28046136443367
// MI455X (gfx1250) — compile-verified, matrix operand order checked by reading, not run on hardware
//
#include <hip/hip_runtime.h>
#include <hip/hip_bf16.h>

// ---------------- problem constants ----------------
constexpr int NN   = 50000;   // nodes
constexpr int FF   = 64;      // features
constexpr int DEG  = 16;      // edges per node (rows are repeat(arange(N),16))
constexpr int HID  = 256;
constexpr int NCLS = 16;
constexpr int N0   = 1351;
constexpr int N1   = 37;
constexpr int CC0P = ((N0 + 15) / 16) * 16;   // 1360
constexpr int CC1P = ((N1 + 15) / 16) * 16;   // 48
constexpr int CC2P = 16;

typedef __attribute__((ext_vector_type(16))) __bf16 v16bf;
typedef __attribute__((ext_vector_type(8)))  __bf16 v8bf;
typedef __attribute__((ext_vector_type(8)))  float  v8f;
typedef __attribute__((ext_vector_type(16))) float  v16f;

__device__ inline __bf16 f2bf(float f) { return (__bf16)f; }   // native RNE cvt

// WMMA 16-bit fragment layout (wave32): lane holds M (A) / N (B) = lane%16,
// and K elements {kk+koff .. +7} U {kk+koff+16 .. +23}, koff = (lane>=16)*8.
// Both halves are contiguous -> two 16B LDS loads / two 32B global loads.
__device__ inline v16bf ldfragBF(const __bf16* p) {
    v8bf lo = *(const v8bf*)p;
    v8bf hi = *(const v8bf*)(p + 16);
    return __builtin_shufflevector(lo, hi, 0, 1, 2, 3, 4, 5, 6, 7,
                                   8, 9, 10, 11, 12, 13, 14, 15);
}
__device__ inline v16bf ldfragF32(const float* p) {
    v8f lo = *(const v8f*)p;
    v8f hi = *(const v8f*)(p + 16);
    v16f a = __builtin_shufflevector(lo, hi, 0, 1, 2, 3, 4, 5, 6, 7,
                                     8, 9, 10, 11, 12, 13, 14, 15);
    return __builtin_convertvector(a, v16bf);
}

// ---------------- BatchNorm ----------------
__global__ void k_bn_stats(const float* __restrict__ x, float* __restrict__ mu,
                           float* __restrict__ var) {
    int f = blockIdx.x;
    float s = 0.f, s2 = 0.f;
    for (int i = threadIdx.x; i < NN; i += blockDim.x) {
        float v = x[i * FF + f];
        s += v; s2 += v * v;
    }
    __shared__ float rs[256], rq[256];
    rs[threadIdx.x] = s; rq[threadIdx.x] = s2;
    __syncthreads();
    for (int o = 128; o > 0; o >>= 1) {
        if ((int)threadIdx.x < o) { rs[threadIdx.x] += rs[threadIdx.x + o]; rq[threadIdx.x] += rq[threadIdx.x + o]; }
        __syncthreads();
    }
    if (threadIdx.x == 0) {
        float m = rs[0] / (float)NN;
        mu[f] = m;
        var[f] = rq[0] / (float)NN - m * m;
    }
}

__global__ void k_bn_apply(const float* __restrict__ x, const float* __restrict__ mu,
                           const float* __restrict__ var, const float* __restrict__ g,
                           const float* __restrict__ b, float* __restrict__ xn) {
    int i = blockIdx.x * blockDim.x + threadIdx.x;
    if (i >= NN * FF) return;
    int f = i % FF;
    xn[i] = (x[i] - mu[f]) * rsqrtf(var[f] + 1e-5f) * g[f] + b[f];
}

__global__ void k_zero(float* __restrict__ p, int n) {
    int i = blockIdx.x * blockDim.x + threadIdx.x;
    if (i < n) p[i] = 0.f;
}

// ---------------- ||C_j||^2 (padded with +inf past nc) ----------------
__global__ void k_rownorm(const float* __restrict__ C, float* __restrict__ cc, int nc) {
    int j = blockIdx.x;
    if (j >= nc) { if (threadIdx.x == 0) cc[j] = 3.0e38f; return; }
    __shared__ float rs[256];
    float v = C[j * HID + threadIdx.x];
    rs[threadIdx.x] = v * v;
    __syncthreads();
    for (int o = 128; o > 0; o >>= 1) {
        if ((int)threadIdx.x < o) rs[threadIdx.x] += rs[threadIdx.x + o];
        __syncthreads();
    }
    if (threadIdx.x == 0) cc[j] = rs[0];
}

// ---------------- fused projection + argmin (WMMA bf16) ----------------
// One wave per 16-row tile. h = xc @ W into LDS (bf16), then sweep 16-wide
// cluster tiles: d = ||h||^2 - 2 h.C^T + ||C||^2, running per-row argmin.
__global__ __launch_bounds__(32)
void k_pool_assign(const float* __restrict__ xc, const float* __restrict__ W,
                   const float* __restrict__ C, const float* __restrict__ cc,
                   int n, int nc, int* __restrict__ assign) {
    __shared__ __bf16 wT[HID * FF];      // W transposed, out-major: wT[o][k]
    __shared__ __bf16 sh[16 * HID];      // h tile (bf16, row-major)
    __shared__ float  hh[16];
    __shared__ float  dt[16 * 16];

    const int lane = threadIdx.x;
    const int row  = lane & 15;
    const int half = (lane >> 4) & 1;
    const int koff = half * 8;
    const int r0   = blockIdx.x * 16;
    int arow = r0 + row; if (arow >= n) arow = n - 1;

    // stage W transposed so B-fragments are contiguous-K LDS b128 loads
    for (int i = lane; i < HID * FF; i += 32) {
        int o = i / FF, k = i % FF;
        wT[i] = f2bf(W[k * HID + o]);
    }
    __syncthreads();

    // h tile: 16 column tiles, K = FF = 64 (2 k-steps of 32)
    for (int ct = 0; ct < HID / 16; ++ct) {
        v8f acc = {};
        for (int kk = 0; kk < FF; kk += 32) {
            v16bf av = ldfragF32(&xc[arow * FF + kk + koff]);
            v16bf bv = ldfragBF(&wT[(ct * 16 + row) * FF + kk + koff]);
            acc = __builtin_amdgcn_wmma_f32_16x16x32_bf16(false, av, false, bv,
                                                          (short)0, acc, false, false);
        }
        #pragma unroll
        for (int v = 0; v < 8; ++v)
            sh[(v + 8 * half) * HID + ct * 16 + row] = f2bf(acc[v]);
    }
    __syncthreads();

    if (lane < 16) {
        float s = 0.f;
        for (int k = 0; k < HID; ++k) { float v = (float)sh[lane * HID + k]; s += v * v; }
        hh[lane] = s;
    }
    __syncthreads();

    float bestd = 3.0e38f; int bestj = 0;
    const int ntiles = (nc + 15) / 16;
    for (int jt = 0; jt < ntiles; ++jt) {
        int col = jt * 16 + row; if (col >= nc) col = nc - 1;
        v8f acc = {};
        for (int kk = 0; kk < HID; kk += 32) {
            v16bf av = ldfragBF(&sh[row * HID + kk + koff]);
            v16bf bv = ldfragF32(&C[col * HID + kk + koff]);
            acc = __builtin_amdgcn_wmma_f32_16x16x32_bf16(false, av, false, bv,
                                                          (short)0, acc, false, false);
        }
        #pragma unroll
        for (int v = 0; v < 8; ++v) {
            int M = v + 8 * half;
            dt[M * 16 + row] = hh[M] - 2.0f * acc[v] + cc[jt * 16 + row];
        }
        __syncthreads();
        if (lane < 16) {
            #pragma unroll
            for (int j = 0; j < 16; ++j) {
                float d = dt[lane * 16 + j];
                if (d < bestd) { bestd = d; bestj = jt * 16 + j; }
            }
        }
        __syncthreads();
    }
    if (lane < 16 && (r0 + lane) < n) assign[r0 + lane] = bestj;
}

// ---------------- mean pooling (scatter-add + divide) ----------------
__global__ void k_pool_scatter(const float* __restrict__ xc, const int* __restrict__ a,
                               float* __restrict__ psum, float* __restrict__ cnt, int n) {
    int i = blockIdx.x * blockDim.x + threadIdx.x;
    if (i >= n * FF) return;
    int row = i / FF, f = i % FF;
    int c = a[row];
    atomicAdd(&psum[c * FF + f], xc[i]);
    if (f == 0) atomicAdd(&cnt[c], 1.0f);
}

__global__ void k_pool_div(float* __restrict__ p, const float* __restrict__ cnt, int nc) {
    int i = blockIdx.x * blockDim.x + threadIdx.x;
    if (i >= nc * FF) return;
    p[i] /= fmaxf(cnt[i / FF], 1.0f);
}

__global__ void k_corr(const float* __restrict__ xc, const float* __restrict__ p,
                       const int* __restrict__ a, float* __restrict__ corr, int n) {
    int i = blockIdx.x * blockDim.x + threadIdx.x;
    if (i >= n * FF) return;
    int row = i / FF, f = i % FF;
    float d = xc[i] - p[a[row] * FF + f];
    corr[i] = d * d;
}

__global__ void k_loss(const float* __restrict__ c1, const float* __restrict__ c2,
                       float* __restrict__ loss) {
    __shared__ float rs[256];
    float s = 0.f;
    for (int i = threadIdx.x; i < N0 * FF; i += 256) s += c1[i];
    float acc = s / (float)N0;
    s = 0.f;
    for (int i = threadIdx.x; i < N1 * FF; i += 256) s += c2[i];
    acc += s / (float)N1;
    rs[threadIdx.x] = acc;
    __syncthreads();
    for (int o = 128; o > 0; o >>= 1) {
        if ((int)threadIdx.x < o) rs[threadIdx.x] += rs[threadIdx.x + o];
        __syncthreads();
    }
    if (threadIdx.x == 0) loss[0] = rs[0] / (float)(FF * 2);
}

// gsq[i,f] = (2*sigmoid(corr_next[a_next[i],f] - corr[i,f]))^2
__global__ void k_gain(const float* __restrict__ corr_next, const int* __restrict__ a_next,
                       const float* __restrict__ corr, float* __restrict__ gsq, int n) {
    int i = blockIdx.x * blockDim.x + threadIdx.x;
    if (i >= n * FF) return;
    int row = i / FF, f = i % FF;
    float z = corr_next[a_next[row] * FF + f] - corr[i];
    float g = 2.0f / (1.0f + __expf(-z));
    gsq[i] = g * g;
}

__global__ void k_compose(const int* __restrict__ a0, const int* __restrict__ a1,
                          int* __restrict__ cl1) {
    int i = blockIdx.x * blockDim.x + threadIdx.x;
    if (i < NN) cl1[i] = a1[a0[i]];
}

// ---------------- fused per-row softmax + SpMM propagation ----------------
// Row r owns edges 16r..16r+15; 64 threads/row (one per feature), 4 rows/block.
__global__ __launch_bounds__(256)
void k_prop(const float* __restrict__ xin, const float* __restrict__ x0,
            const float* __restrict__ adj_vals, const int* __restrict__ adj_cols,
            const int* __restrict__ cl0, const int* __restrict__ cl1,
            const float* __restrict__ g0sq, const float* __restrict__ g1sq,
            float* __restrict__ xout) {
    __shared__ int   scol[4][DEG];
    __shared__ float sadj[4][DEG];
    __shared__ int   sf0[4][DEG];
    __shared__ int   sf1[4][DEG];
    const int lr = threadIdx.x / FF;
    const int f  = threadIdx.x % FF;
    const int r  = blockIdx.x * 4 + lr;
    if (r < NN && f < DEG) {
        int e = r * DEG + f;
        int c = adj_cols[e];
        scol[lr][f] = c;
        sadj[lr][f] = adj_vals[e];
        sf0[lr][f]  = (cl0[c] == cl0[r]);
        sf1[lr][f]  = (cl1[c] == cl1[r]);
    }
    __syncthreads();
    if (r >= NN) return;
    const float g0v = g0sq[cl0[r] * FF + f];
    const float g1v = g1sq[cl1[r] * FF + f];
    float vals[DEG];
    float m = -3.0e38f;
    #pragma unroll
    for (int e = 0; e < DEG; ++e) {
        float v = sadj[lr][e] + (sf0[lr][e] ? g0v : 0.f) + (sf1[lr][e] ? g1v : 0.f);
        vals[e] = v;
        m = fmaxf(m, v);
    }
    float s = 0.f, acc = 0.f;
    #pragma unroll
    for (int e = 0; e < DEG; ++e) {
        float ev = __expf(vals[e] - m);
        s += ev;
        acc += ev * xin[scol[lr][e] * FF + f];
    }
    xout[r * FF + f] = acc / s + x0[r * FF + f];
}

// ---------------- fused MLP head, weights staged in LDS (bf16) ----------------
constexpr size_t SMEM_MLP =
    (size_t)(HID * 2 * FF + HID * HID + NCLS * HID) * 2   // wT1, wT2, wT3 (bf16)
  + (size_t)(HID + HID + NCLS) * 4                        // biases (f32)
  + (size_t)(16 * 2 * FF + 16 * HID + 16 * HID) * 2       // xcat, h1, h2 (bf16)
  + (size_t)(16 * 16) * 4;                                // logits (f32)

__global__ __launch_bounds__(128)
void k_mlp(const float* __restrict__ xprop, const float* __restrict__ x0,
           const float* __restrict__ mW1, const float* __restrict__ mb1,
           const float* __restrict__ mp1, const float* __restrict__ mW2,
           const float* __restrict__ mb2, const float* __restrict__ mp2,
           const float* __restrict__ mW3, const float* __restrict__ mb3,
           float* __restrict__ out) {
    extern __shared__ char smem[];
    __bf16* wT1 = (__bf16*)smem;                   // [HID][2F]  (out-major)
    __bf16* wT2 = wT1 + HID * 2 * FF;              // [HID][HID]
    __bf16* wT3 = wT2 + HID * HID;                 // [NCLS][HID]
    float*  sb1 = (float*)(wT3 + NCLS * HID);
    float*  sb2 = sb1 + HID;
    float*  sb3 = sb2 + HID;
    __bf16* xcat = (__bf16*)(sb3 + NCLS);          // [16][2F]
    __bf16* h1   = xcat + 16 * 2 * FF;             // [16][HID]
    __bf16* h2   = h1 + 16 * HID;                  // [16][HID]
    float*  lg   = (float*)(h2 + 16 * HID);        // [16][16]

    const int tid = threadIdx.x;
    for (int i = tid; i < HID * 2 * FF; i += 128) {
        int o = i / (2 * FF), k = i % (2 * FF);
        wT1[i] = f2bf(mW1[k * HID + o]);
    }
    for (int i = tid; i < HID * HID; i += 128) {
        int o = i / HID, k = i % HID;
        wT2[i] = f2bf(mW2[k * HID + o]);
    }
    for (int i = tid; i < NCLS * HID; i += 128) {
        int o = i / HID, k = i % HID;
        wT3[i] = f2bf(mW3[k * NCLS + o]);
    }
    for (int i = tid; i < HID; i += 128) { sb1[i] = mb1[i]; sb2[i] = mb2[i]; }
    if (tid < NCLS) sb3[tid] = mb3[tid];
    const float pr1 = mp1[0], pr2 = mp2[0];
    __syncthreads();

    const int wave = tid >> 5;
    const int lane = tid & 31;
    const int row  = lane & 15;
    const int half = lane >> 4;
    const int koff = half * 8;

    for (int t = blockIdx.x; t < NN / 16; t += gridDim.x) {
        const int r0 = t * 16;
        for (int i = tid; i < 16 * 2 * FF; i += 128) {
            int rr = i / (2 * FF), c = i % (2 * FF);
            int gr = r0 + rr; if (gr >= NN) gr = NN - 1;
            float v = (c < FF) ? xprop[gr * FF + c] : x0[gr * FF + (c - FF)];
            xcat[i] = f2bf(v);
        }
        __syncthreads();

        // layer 1: [16,128] @ [128,256] + b1, PReLU
        for (int q = 0; q < 4; ++q) {
            int ct = wave * 4 + q;
            v8f acc = {};
            for (int kk = 0; kk < 2 * FF; kk += 32) {
                v16bf av = ldfragBF(&xcat[row * (2 * FF) + kk + koff]);
                v16bf bv = ldfragBF(&wT1[(ct * 16 + row) * (2 * FF) + kk + koff]);
                acc = __builtin_amdgcn_wmma_f32_16x16x32_bf16(false, av, false, bv,
                                                              (short)0, acc, false, false);
            }
            #pragma unroll
            for (int v = 0; v < 8; ++v) {
                float h = acc[v] + sb1[ct * 16 + row];
                h = (h >= 0.f) ? h : pr1 * h;
                h1[(v + 8 * half) * HID + ct * 16 + row] = f2bf(h);
            }
        }
        __syncthreads();

        // layer 2: [16,256] @ [256,256] + b2, PReLU
        for (int q = 0; q < 4; ++q) {
            int ct = wave * 4 + q;
            v8f acc = {};
            for (int kk = 0; kk < HID; kk += 32) {
                v16bf av = ldfragBF(&h1[row * HID + kk + koff]);
                v16bf bv = ldfragBF(&wT2[(ct * 16 + row) * HID + kk + koff]);
                acc = __builtin_amdgcn_wmma_f32_16x16x32_bf16(false, av, false, bv,
                                                              (short)0, acc, false, false);
            }
            #pragma unroll
            for (int v = 0; v < 8; ++v) {
                float h = acc[v] + sb2[ct * 16 + row];
                h = (h >= 0.f) ? h : pr2 * h;
                h2[(v + 8 * half) * HID + ct * 16 + row] = f2bf(h);
            }
        }
        __syncthreads();

        // layer 3: [16,256] @ [256,16] + b3 (wave 0 only)
        if (wave == 0) {
            v8f acc = {};
            for (int kk = 0; kk < HID; kk += 32) {
                v16bf av = ldfragBF(&h2[row * HID + kk + koff]);
                v16bf bv = ldfragBF(&wT3[row * HID + kk + koff]);
                acc = __builtin_amdgcn_wmma_f32_16x16x32_bf16(false, av, false, bv,
                                                              (short)0, acc, false, false);
            }
            #pragma unroll
            for (int v = 0; v < 8; ++v)
                lg[(v + 8 * half) * 16 + row] = acc[v] + sb3[row];
        }
        __syncthreads();

        // log_softmax per row
        if (tid < 16 && (r0 + tid) < NN) {
            float mx = -3.0e38f;
            #pragma unroll
            for (int j = 0; j < NCLS; ++j) mx = fmaxf(mx, lg[tid * 16 + j]);
            float s = 0.f;
            #pragma unroll
            for (int j = 0; j < NCLS; ++j) s += __expf(lg[tid * 16 + j] - mx);
            float lse = mx + __logf(s);
            #pragma unroll
            for (int j = 0; j < NCLS; ++j)
                out[(r0 + tid) * NCLS + j] = lg[tid * 16 + j] - lse;
        }
        __syncthreads();
    }
}

// ---------------- driver ----------------
extern "C" void kernel_launch(void* const* d_in, const int* in_sizes, int n_in,
                              void* d_out, int out_size, void* d_ws, size_t ws_size,
                              hipStream_t stream) {
    (void)in_sizes; (void)n_in; (void)out_size; (void)ws_size;
    const float* x        = (const float*)d_in[0];
    const float* x_cov    = (const float*)d_in[1];
    const float* adj_vals = (const float*)d_in[2];
    const float* fcorr    = (const float*)d_in[3];
    const float* bn_g     = (const float*)d_in[4];
    const float* bn_b     = (const float*)d_in[5];
    const float* W0       = (const float*)d_in[6];
    const float* C0       = (const float*)d_in[7];
    const float* W1       = (const float*)d_in[8];
    const float* C1       = (const float*)d_in[9];
    const float* W2       = (const float*)d_in[10];
    const float* C2       = (const float*)d_in[11];
    const float* mW1      = (const float*)d_in[12];
    const float* mb1      = (const float*)d_in[13];
    const float* mp1      = (const float*)d_in[14];
    const float* mW2      = (const float*)d_in[15];
    const float* mb2      = (const float*)d_in[16];
    const float* mp2      = (const float*)d_in[17];
    const float* mW3      = (const float*)d_in[18];
    const float* mb3      = (const float*)d_in[19];
    const int*   adj_cols = (const int*)d_in[21];
    float* out = (float*)d_out;

    // workspace carve-up (all 4-byte typed)
    float* w = (float*)d_ws;
    float* xn  = w; w += NN * FF;
    float* xp  = w; w += NN * FF;
    float* xq  = w; w += NN * FF;
    float* mu  = w; w += FF;
    float* var = w; w += FF;
    float* p0  = w; w += N0 * FF;     // p0..cnt2 contiguous -> single zero pass
    float* p1  = w; w += N1 * FF;
    float* p2  = w; w += FF;
    float* cnt0 = w; w += N0;
    float* cnt1 = w; w += N1;
    float* cnt2 = w; w += 1;
    float* corr1 = w; w += N0 * FF;
    float* corr2 = w; w += N1 * FF;
    float* g0sq  = w; w += N0 * FF;
    float* g1sq  = w; w += N1 * FF;
    float* cc0 = w; w += CC0P;
    float* cc1 = w; w += CC1P;
    float* cc2 = w; w += CC2P;
    int* a0  = (int*)w; w += NN;
    int* a1  = (int*)w; w += N0;
    int* a2  = (int*)w; w += N1;
    int* cl1 = (int*)w; w += NN;

    const int NF = NN * FF;
    const int zeroN = N0 * FF + N1 * FF + FF + N0 + N1 + 1;

    // BatchNorm
    k_bn_stats<<<FF, 256, 0, stream>>>(x, mu, var);
    k_bn_apply<<<(NF + 255) / 256, 256, 0, stream>>>(x, mu, var, bn_g, bn_b, xn);

    // pooling hierarchy
    k_zero<<<(zeroN + 255) / 256, 256, 0, stream>>>(p0, zeroN);
    k_rownorm<<<CC0P, 256, 0, stream>>>(C0, cc0, N0);
    k_rownorm<<<CC1P, 256, 0, stream>>>(C1, cc1, N1);
    k_rownorm<<<CC2P, 256, 0, stream>>>(C2, cc2, 1);

    k_pool_assign<<<(NN + 15) / 16, 32, 0, stream>>>(x_cov, W0, C0, cc0, NN, N0, a0);
    k_pool_scatter<<<(NF + 255) / 256, 256, 0, stream>>>(x_cov, a0, p0, cnt0, NN);
    k_pool_div<<<(N0 * FF + 255) / 256, 256, 0, stream>>>(p0, cnt0, N0);

    k_pool_assign<<<(N0 + 15) / 16, 32, 0, stream>>>(p0, W1, C1, cc1, N0, N1, a1);
    k_pool_scatter<<<(N0 * FF + 255) / 256, 256, 0, stream>>>(p0, a1, p1, cnt1, N0);
    k_pool_div<<<(N1 * FF + 255) / 256, 256, 0, stream>>>(p1, cnt1, N1);
    k_corr<<<(N0 * FF + 255) / 256, 256, 0, stream>>>(p0, p1, a1, corr1, N0);

    k_pool_assign<<<(N1 + 15) / 16, 32, 0, stream>>>(p1, W2, C2, cc2, N1, 1, a2);
    k_pool_scatter<<<(N1 * FF + 255) / 256, 256, 0, stream>>>(p1, a2, p2, cnt2, N1);
    k_pool_div<<<(FF + 255) / 256, 256, 0, stream>>>(p2, cnt2, 1);
    k_corr<<<(N1 * FF + 255) / 256, 256, 0, stream>>>(p1, p2, a2, corr2, N1);

    // loss -> last output element
    k_loss<<<1, 256, 0, stream>>>(corr1, corr2, out + (size_t)NN * NCLS);

    // gains + composed cluster ids
    k_gain<<<(N0 * FF + 255) / 256, 256, 0, stream>>>(corr2, a1, corr1, g0sq, N0);
    k_gain<<<(N1 * FF + 255) / 256, 256, 0, stream>>>(fcorr, a2, corr2, g1sq, N1);
    k_compose<<<(NN + 255) / 256, 256, 0, stream>>>(a0, a1, cl1);

    // K=2 fused softmax+propagation steps
    k_prop<<<(NN + 3) / 4, 256, 0, stream>>>(xn, xn, adj_vals, adj_cols, a0, cl1, g0sq, g1sq, xp);
    k_prop<<<(NN + 3) / 4, 256, 0, stream>>>(xp, xn, adj_vals, adj_cols, a0, cl1, g0sq, g1sq, xq);

    // fused MLP head with all weights in LDS (bf16, ~223 KB dynamic LDS)
    (void)hipFuncSetAttribute((const void*)k_mlp,
                              hipFuncAttributeMaxDynamicSharedMemorySize, (int)SMEM_MLP);
    k_mlp<<<240, 128, SMEM_MLP, stream>>>(xq, xn, mW1, mb1, mp1, mW2, mb2, mp2, mW3, mb3, out);
}
